// Head_24094766531333
// MI455X (gfx1250) — matrix operand order repaired and checked by reading, not compiled, not run on hardware
//
#include <hip/hip_runtime.h>
#include <hip/hip_bf16.h>
#include <math.h>

typedef __attribute__((ext_vector_type(16))) _Float16     v16h;
typedef __attribute__((ext_vector_type(8)))  float        v8f;
typedef __attribute__((ext_vector_type(4)))  unsigned int v4u;

#define TDIM 4096
#define CDIM 1024
#define HDIM 64
#define BDIM 4

// H^-0.5 * log2(e) folded into Q so softmax uses exp2
#define QSCALE 0.18033688011112042f

// ---------------------------------------------------------------------------
// Kernel 1: fused QKV projection.  x[16384,1024] f32 x W[1024,64] f32 (x3)
// -> Q f16 [B*T,64] (pre-scaled), K f16 [B*T,64], Vt f16 [B,64,T] (transposed)
// Each wave computes a 16-row tile; W chunks staged transposed in LDS.
// ---------------------------------------------------------------------------
__global__ __launch_bounds__(256) void qkv_proj_kernel(
    const float* __restrict__ x,
    const float* __restrict__ Wq, const float* __restrict__ Wk,
    const float* __restrict__ Wv,
    _Float16* __restrict__ qb, _Float16* __restrict__ kb,
    _Float16* __restrict__ vt)
{
    __shared__ _Float16 wt[3][64][32];   // wt[w][n][k] = W[c*32+k][n] as f16

    const int tid  = threadIdx.x;
    const int wave = tid >> 5;
    const int lane = tid & 31;
    const int g    = lane >> 4;   // half-wave group
    const int ln   = lane & 15;
    const int rowbase = blockIdx.x * 128 + wave * 16;
    const int arow    = rowbase + ln;            // A-matrix row for this lane

    const float* Ws[3] = {Wq, Wk, Wv};

    v8f acc[3][4] = {};   // 3 projections x 4 N-tiles of 16

    for (int c = 0; c < 32; ++c) {
        __syncthreads();  // protect WAR on wt from previous iteration
        // ---- stage transposed W chunk (f32 -> f16) cooperatively ----
        #pragma unroll
        for (int w = 0; w < 3; ++w) {
            const float* W = Ws[w];
            #pragma unroll
            for (int j = 0; j < 4; ++j) {
                int idx = j * 256 + tid;       // 0..1023
                int n   = idx & 63;
                int kp  = idx >> 6;            // k-pair 0..15
                float f0 = W[(c * 32 + kp * 2 + 0) * 64 + n];
                float f1 = W[(c * 32 + kp * 2 + 1) * 64 + n];
                wt[w][n][kp * 2 + 0] = (_Float16)f0;
                wt[w][n][kp * 2 + 1] = (_Float16)f1;
            }
        }
        __syncthreads();

        // ---- A tile from x (ISA 16-bit A layout: runs of 8 halves) ----
        union { v16h v; _Float16 h[16]; } a;
        const float* xr = x + (size_t)arow * CDIM + c * 32;
        #pragma unroll
        for (int e = 0; e < 8; ++e) a.h[e]     = (_Float16)xr[g * 8 + e];
        #pragma unroll
        for (int e = 0; e < 8; ++e) a.h[8 + e] = (_Float16)xr[16 + g * 8 + e];

        // ---- 12 WMMAs: 3 matrices x 4 N-tiles ----
        #pragma unroll
        for (int w = 0; w < 3; ++w) {
            #pragma unroll
            for (int t = 0; t < 4; ++t) {
                const _Float16* bp = &wt[w][t * 16 + ln][g * 8]; // two 16B runs: k=8g.., 16+8g..
                union { v16h v; v4u u[2]; } b;
                b.u[0] = *(const v4u*)(bp);
                b.u[1] = *(const v4u*)(bp + 16);
                acc[w][t] = __builtin_amdgcn_wmma_f32_16x16x32_f16(
                    false, a.v, false, b.v, (short)0, acc[w][t], false, false);
            }
        }
    }

    // ---- store Q (scaled), K, Vt (transposed) as f16 ----
    #pragma unroll
    for (int t = 0; t < 4; ++t) {
        #pragma unroll
        for (int r = 0; r < 8; ++r) {
            int row = rowbase + r + 8 * g;   // C-layout: M = r + 8*group
            int n   = t * 16 + ln;           // N = lane&15
            qb[(size_t)row * HDIM + n] = (_Float16)(acc[0][t][r] * QSCALE);
            kb[(size_t)row * HDIM + n] = (_Float16)(acc[1][t][r]);
            int b  = row >> 12;              // T = 4096
            int tt = row & (TDIM - 1);
            vt[((size_t)b * HDIM + n) * TDIM + tt] = (_Float16)(acc[2][t][r]);
        }
    }
}

// ---------------------------------------------------------------------------
// Kernel 2: causal flash attention.  Each wave = one 16-query tile.
// Loops over 32-key chunks; online softmax in exp2 domain.
// ---------------------------------------------------------------------------
__global__ __launch_bounds__(256) void attn_kernel(
    const _Float16* __restrict__ qbuf, const _Float16* __restrict__ kbuf,
    const _Float16* __restrict__ vtbuf, float* __restrict__ out)
{
    __shared__ _Float16 pst[8][16][32];   // per-wave P staging (C->A relayout)

    const int tid  = threadIdx.x;
    const int wave = tid >> 5;
    const int lane = tid & 31;
    const int g    = lane >> 4;
    const int ln   = lane & 15;

    const int bidx  = blockIdx.x >> 5;                  // batch (32 blocks/batch)
    const int qbase = (blockIdx.x & 31) * 128 + wave * 16;
    const int tok0  = bidx * TDIM;

    // ---- load Q A-tiles (head 0..31 and 32..63), contiguous 16B runs ----
    const _Float16* qrow = qbuf + (size_t)(tok0 + qbase + ln) * HDIM;
    v16h a_lo, a_hi;
    {
        union { v16h v; v4u u[2]; } t0, t1;
        t0.u[0] = *(const v4u*)(qrow + g * 8);
        t0.u[1] = *(const v4u*)(qrow + 16 + g * 8);
        t1.u[0] = *(const v4u*)(qrow + 32 + g * 8);
        t1.u[1] = *(const v4u*)(qrow + 48 + g * 8);
        a_lo = t0.v; a_hi = t1.v;
    }

    v8f o[4] = {};
    float m[8], l[8];
    #pragma unroll
    for (int r = 0; r < 8; ++r) { m[r] = -1e30f; l[r] = 0.0f; }

    const int  cmax = (qbase + 15) >> 5;
    _Float16*  pw   = &pst[wave][0][0];

    for (int c = 0; c <= cmax; ++c) {
        const int key0 = c * 32;

        if (c < cmax)  // prefetch next K chunk (global_prefetch_b8)
            __builtin_prefetch(kbuf + (size_t)(tok0 + key0 + 32 + lane) * HDIM, 0, 0);

        // ---- S = Q K^T  (two 16-key N-tiles, contraction 64 = 2x32) ----
        v8f s[2];
        #pragma unroll
        for (int t = 0; t < 2; ++t) {
            const _Float16* krow =
                kbuf + (size_t)(tok0 + key0 + t * 16 + ln) * HDIM + g * 8;
            union { v16h v; v4u u[2]; } b0, b1;
            b0.u[0] = *(const v4u*)(krow);
            b0.u[1] = *(const v4u*)(krow + 16);
            b1.u[0] = *(const v4u*)(krow + 32);
            b1.u[1] = *(const v4u*)(krow + 48);
            v8f z = {};
            s[t] = __builtin_amdgcn_wmma_f32_16x16x32_f16(
                false, a_lo, false, b0.v, (short)0, z, false, false);
            s[t] = __builtin_amdgcn_wmma_f32_16x16x32_f16(
                false, a_hi, false, b1.v, (short)0, s[t], false, false);
        }

        // ---- causal mask (only last chunk can cross the diagonal) ----
        if (c == cmax) {
            #pragma unroll
            for (int t = 0; t < 2; ++t) {
                int key = key0 + t * 16 + ln;
                #pragma unroll
                for (int r = 0; r < 8; ++r) {
                    int q = qbase + r + 8 * g;
                    if (key > q) s[t][r] = -1e30f;
                }
            }
        }

        // ---- online softmax (exp2 domain; scale folded into Q) ----
        float corr[8], psum[8];
        #pragma unroll
        for (int r = 0; r < 8; ++r) {
            float v = fmaxf(s[0][r], s[1][r]);
            v = fmaxf(v, __shfl_xor(v, 1));
            v = fmaxf(v, __shfl_xor(v, 2));
            v = fmaxf(v, __shfl_xor(v, 4));
            v = fmaxf(v, __shfl_xor(v, 8));   // row max across 16-lane group
            float mn = fmaxf(m[r], v);
            corr[r]  = exp2f(m[r] - mn);
            m[r]     = mn;
            float p0 = exp2f(s[0][r] - mn);
            float p1 = exp2f(s[1][r] - mn);
            pw[(r + 8 * g) * 32 + ln]      = (_Float16)p0;  // row-major P in LDS
            pw[(r + 8 * g) * 32 + 16 + ln] = (_Float16)p1;
            psum[r] = p0 + p1;
        }
        #pragma unroll
        for (int r = 0; r < 8; ++r) {
            float v = psum[r];
            v += __shfl_xor(v, 1);
            v += __shfl_xor(v, 2);
            v += __shfl_xor(v, 4);
            v += __shfl_xor(v, 8);
            l[r] = l[r] * corr[r] + v;
            #pragma unroll
            for (int t = 0; t < 4; ++t) o[t][r] *= corr[r];
        }

        // same-wave DS store->load ordering (no workgroup barrier: waves have
        // different trip counts). DS is in-order per wave; wait + compiler fence.
        asm volatile("s_wait_dscnt 0" ::: "memory");

        // ---- reload P in A-matrix layout (swizzled contiguous 16B runs) ----
        v16h a_p;
        {
            const _Float16* pr = pw + ln * 32;
            union { v16h v; v4u u[2]; } t0;
            t0.u[0] = *(const v4u*)(pr + g * 16);
            t0.u[1] = *(const v4u*)(pr + g * 16 + 8);
            a_p = t0.v;
        }

        // ---- O += P @ V  (V pre-transposed: contiguous 32B per lane) ----
        #pragma unroll
        for (int t = 0; t < 4; ++t) {
            const _Float16* vr =
                vtbuf + ((size_t)bidx * HDIM + t * 16 + ln) * TDIM + key0 + g * 16;
            union { v16h v; v4u u[2]; } bv;
            bv.u[0] = *(const v4u*)(vr);
            bv.u[1] = *(const v4u*)(vr + 8);
            o[t] = __builtin_amdgcn_wmma_f32_16x16x32_f16(
                false, a_p, false, bv.v, (short)0, o[t], false, false);
        }
    }

    // ---- normalize and store f32 output ----
    #pragma unroll
    for (int t = 0; t < 4; ++t) {
        #pragma unroll
        for (int r = 0; r < 8; ++r) {
            int row = qbase + r + 8 * g;
            out[((size_t)tok0 + row) * HDIM + t * 16 + ln] = o[t][r] / l[r];
        }
    }
}

// ---------------------------------------------------------------------------
extern "C" void kernel_launch(void* const* d_in, const int* in_sizes, int n_in,
                              void* d_out, int out_size, void* d_ws, size_t ws_size,
                              hipStream_t stream) {
    const float* x  = (const float*)d_in[0];
    const float* Wq = (const float*)d_in[1];
    const float* Wk = (const float*)d_in[2];
    const float* Wv = (const float*)d_in[3];
    float* out = (float*)d_out;

    const size_t NT = (size_t)BDIM * TDIM * HDIM;   // 1M elements
    _Float16* qb = (_Float16*)d_ws;                 // 2 MB
    _Float16* kb = qb + NT;                         // 2 MB
    _Float16* vt = kb + NT;                         // 2 MB (transposed V)

    // 128 blocks x 256 threads: each block = 128 rows of B*T = 16384
    qkv_proj_kernel<<<128, 256, 0, stream>>>(x, Wq, Wk, Wv, qb, kb, vt);
    // 128 blocks: 8 waves/block x 16 queries/wave = 128 queries per block
    attn_kernel<<<128, 256, 0, stream>>>(qb, kb, vt, out);
}
